// TensorFieldNetworkModule_14018773254608
// MI455X (gfx1250) — matrix operand order repaired and checked by reading, not compiled, not run on hardware
//
#include <hip/hip_runtime.h>
#include <hip/hip_bf16.h>
#include <math.h>

#define N_ATOMS 768
#define C 128
#define A_TYPES 8
#define SL 11           // accumulator slots: [a0, o0, b1.xyz, a1.xyz, o1.xyz]
#define W2P 136         // padded row stride (shorts) for s_w2: bank-conflict-free B loads

typedef __attribute__((ext_vector_type(16))) __bf16 v16bf;
typedef __attribute__((ext_vector_type(8)))  float  v8f;

union ABFrag { v16bf v; unsigned short s[16]; uint4 q[2]; };

__device__ __forceinline__ unsigned short f2bf(float x) {
  unsigned int u = __float_as_uint(x);
  unsigned int r = (u + 0x7FFFu + ((u >> 16) & 1u)) >> 16;   // RNE
  return (unsigned short)r;
}
__device__ __forceinline__ float elu1(float z) { return z > 0.f ? z : (expf(z) - 1.f); }

// ---------------- embedding: e[a,f] = one_hot @ embed_W.T + b ----------------
__global__ void embed_kernel(const float* __restrict__ oneh, const float* __restrict__ W,
                             const float* __restrict__ b, float* __restrict__ e) {
  int idx = blockIdx.x * blockDim.x + threadIdx.x;
  if (idx >= N_ATOMS * C) return;
  int a = idx / C, f = idx % C;
  float s = b[f];
#pragma unroll
  for (int t = 0; t < A_TYPES; ++t) s += oneh[a * A_TYPES + t] * W[f * A_TYPES + t];
  e[idx] = s;
}

// ------------- W2 transpose + bf16 convert: W2T[i][d][c] = bf16(Rw2[i][c][d]) -------------
__global__ void w2t_kernel(const float* __restrict__ Rw2, unsigned short* __restrict__ W2T) {
  int idx = blockIdx.x * blockDim.x + threadIdx.x;
  if (idx >= 12 * C * C) return;
  int i = idx / (C * C); int rem = idx % (C * C);
  int d = rem / C, c = rem % C;
  W2T[i * C * C + d * C + c] = f2bf(Rw2[i * C * C + c * C + d]);
}

// ---------------- fused pair kernel: geometry + radial MLP (WMMA) + contraction ----------------
// KIND==0: layer0 (2 filters on e), KIND==1: full layer (5 filters on x0/x1)
template <int KIND>
__global__ __launch_bounds__(256)
void pair_kernel(const float* __restrict__ r,
                 const float* __restrict__ x0in, const float* __restrict__ x1in,
                 const float* __restrict__ Rw1, const float* __restrict__ Rb1,
                 const unsigned short* __restrict__ W2T, const float* __restrict__ Rb2,
                 float* __restrict__ P, int ri0) {
  __shared__ float s_rbf[256 * 4];                 //  4 KB
  __shared__ float s_u[256 * 3];                   //  3 KB
  __shared__ float s_x0[16 * C];                   //  8 KB
  __shared__ float s_x1[16 * C * 3];               // 24 KB
  __shared__ __align__(16) unsigned short s_w2[C * W2P]; // 34 KB (bf16, padded [d][c])
  __shared__ float s_w1[4 * C];                    //  2 KB
  __shared__ float s_rb1[C];
  __shared__ float s_rb2[C];
  __shared__ float s_acc[16 * C * SL];             // 88 KB

  const int tid  = threadIdx.x;
  const int lane = tid & 31;
  const int wave = tid >> 5;
  const int aBase  = blockIdx.x * 16;
  const int split  = blockIdx.y;
  const int nsplit = gridDim.y;
  const int NF     = KIND ? 5 : 2;
  const int nslots = KIND ? 11 : 4;

  for (int i = tid; i < 16 * C * SL; i += 256) s_acc[i] = 0.f;

  for (int bt = split; bt < N_ATOMS / 16; bt += nsplit) {
    __syncthreads();
    // ---- geometry for 256 pairs (16a x 16b) ----
    {
      int p = tid;
      int aL = p >> 4, bL = p & 15;
      int aG = aBase + aL, bG = bt * 16 + bL;
      float dx = r[aG * 3 + 0] - r[bG * 3 + 0];
      float dy = r[aG * 3 + 1] - r[bG * 3 + 1];
      float dz = r[aG * 3 + 2] - r[bG * 3 + 2];
      float s2 = dx * dx + dy * dy + dz * dz;
      float d  = sqrtf(fmaxf(s2, 1e-12f));
      float iv = 1.f / (d + 1e-8f);
      s_u[p * 3 + 0] = dx * iv; s_u[p * 3 + 1] = dy * iv; s_u[p * 3 + 2] = dz * iv;
#pragma unroll
      for (int k = 0; k < 4; ++k) {
        float t = d - 0.833333333f * (float)k;     // centers linspace(0,2.5,4)
        s_rbf[p * 4 + k] = __expf(-1.6f * t * t);  // gamma = RBF/(hi-lo)
      }
    }
    // ---- stage b-tile features ----
    for (int i = tid; i < 16 * C; i += 256) s_x0[i] = x0in[bt * 16 * C + i];
    if (KIND)
      for (int i = tid; i < 16 * C * 3; i += 256) s_x1[i] = x1in[bt * 16 * C * 3 + i];

#pragma unroll
    for (int f = 0; f < NF; ++f) {
      // compile-time role/slot (filter loop is fully unrolled)
      const int role = (KIND == 0) ? ((f == 0) ? 0 : 1)
                                   : (f == 0 ? 0 : f == 1 ? 1 : f == 2 ? 2 : f == 3 ? 3 : 4);
      const int slot = (KIND == 0) ? ((f == 0) ? 0 : 1)
                                   : (f == 0 ? 0 : f == 1 ? 2 : f == 2 ? 5 : f == 3 ? 1 : 8);
      __syncthreads();
      const int fid = ri0 + f;
      for (int i = tid; i < 4 * C; i += 256) s_w1[i] = Rw1[fid * 4 * C + i];
      for (int i = tid; i < C; i += 256) { s_rb1[i] = Rb1[fid * C + i]; s_rb2[i] = Rb2[fid * C + i]; }
      {
        const unsigned int* src = (const unsigned int*)(W2T + fid * C * C);
        unsigned int* dst = (unsigned int*)s_w2;
        for (int i = tid; i < C * C / 2; i += 256) {
          int row = i / (C / 2), col = i % (C / 2);
          dst[row * (W2P / 2) + col] = src[i];
        }
      }
      __syncthreads();

      // each wave owns two a-rows (tileM); rows of the matmul are the 16 b atoms
      for (int tm = wave * 2; tm < wave * 2 + 2; ++tm) {
        const int bRow = lane & 15;
        const float* rb = &s_rbf[(tm * 16 + bRow) * 4];
        float r0 = rb[0], r1 = rb[1], r2 = rb[2], r3 = rb[3];
        const int hi  = (lane >= 16) ? 16 : 0;
        const int coH = (lane >= 16) ? 16 : 0;

        // hidden layer (K=4 MLP) -> four A-fragments, k = 16*(lane/16) + s (32 VGPRs)
        ABFrag afr[4];
#pragma unroll
        for (int kk = 0; kk < 4; ++kk) {
          int kb = kk * 32 + hi;
#pragma unroll
          for (int j = 0; j < 8; ++j) {
            int c = kb + j;
            float h = s_rb1[c] + r0 * s_w1[c] + r1 * s_w1[C + c] + r2 * s_w1[2 * C + c] + r3 * s_w1[3 * C + c];
            afr[kk].s[j] = f2bf(fmaxf(h, 0.f));
          }
#pragma unroll
          for (int j = 0; j < 8; ++j) {
            int c = kb + 8 + j;
            float h = s_rb1[c] + r0 * s_w1[c] + r1 * s_w1[C + c] + r2 * s_w1[2 * C + c] + r3 * s_w1[3 * C + c];
            afr[kk].s[8 + j] = f2bf(fmaxf(h, 0.f));
          }
        }

        const int mBase = (lane >= 16) ? 8 : 0;
        const float* uRow = &s_u[tm * 48];
#pragma unroll 2
        for (int tn = 0; tn < 8; ++tn) {
          int ch = tn * 16 + (lane & 15);
          const unsigned short* wrow = &s_w2[ch * W2P + coH];
          v8f acc;
#pragma unroll
          for (int e2 = 0; e2 < 8; ++e2) acc[e2] = 0.f;
#pragma unroll
          for (int kk = 0; kk < 4; ++kk) {
            ABFrag bf;
            const uint4* bp = (const uint4*)(wrow + kk * 32);
            bf.q[0] = bp[0]; bf.q[1] = bp[1];
            acc = __builtin_amdgcn_wmma_f32_16x16x32_bf16(
                false, afr[kk].v, false, bf.v, (short)0, acc, false, false);
          }

          // ---- epilogue (straight-line; role/slot are constants) ----
          float bias2 = s_rb2[ch];
          float sA = 0.f, vx = 0.f, vy = 0.f, vz = 0.f;
#pragma unroll
          for (int vv = 0; vv < 8; ++vv) {
            int bL = mBase + vv;
            float R = acc[vv] + bias2;
            if (role == 0) {
              sA += R * s_x0[bL * C + ch];
            } else if (role == 1) {
              float xv = R * s_x0[bL * C + ch];
              const float* uu = &uRow[bL * 3];
              vx += xv * uu[0]; vy += xv * uu[1]; vz += xv * uu[2];
            } else if (role == 2) {
              const float* x1 = &s_x1[(bL * C + ch) * 3];
              vx += R * x1[0]; vy += R * x1[1]; vz += R * x1[2];
            } else if (role == 3) {
              const float* uu = &uRow[bL * 3];
              const float* x1 = &s_x1[(bL * C + ch) * 3];
              sA += R * (uu[0] * x1[0] + uu[1] * x1[1] + uu[2] * x1[2]);
            } else {
              const float* uu = &uRow[bL * 3];
              const float* x1 = &s_x1[(bL * C + ch) * 3];
              vx += R * (uu[1] * x1[2] - uu[2] * x1[1]);
              vy += R * (uu[2] * x1[0] - uu[0] * x1[2]);
              vz += R * (uu[0] * x1[1] - uu[1] * x1[0]);
            }
          }
          if (role == 0 || role == 3) {
            sA += __shfl_xor(sA, 16, 32);
            if (lane < 16) s_acc[(tm * C + ch) * SL + slot] += sA;
          } else {
            vx += __shfl_xor(vx, 16, 32);
            vy += __shfl_xor(vy, 16, 32);
            vz += __shfl_xor(vz, 16, 32);
            if (lane < 16) {
              float* Aq = &s_acc[(tm * C + ch) * SL + slot];
              Aq[0] += vx; Aq[1] += vy; Aq[2] += vz;
            }
          }
        }
      }
    }
  }
  __syncthreads();
  // ---- write this split's partial sums ----
  int total = 16 * C * nslots;
  for (int i = tid; i < total; i += 256) {
    int aL  = i / (C * nslots);
    int rem = i % (C * nslots);
    int ch  = rem / nslots;
    int sl  = rem % nslots;
    P[(((long)split * N_ATOMS + aBase + aL) * C + ch) * SL + sl] = s_acc[(aL * C + ch) * SL + sl];
  }
}

// ---------------- self-interaction + nonlinearity (+ final prediction) ----------------
// kind: 0=layer0 (nonlin), 1=layer1 (nonlin), 2=layer2 (linear, writes final outputs)
__global__ __launch_bounds__(128)
void si_kernel(const float* __restrict__ P, int nsplit, int kind,
               const float* __restrict__ w0, const float* __restrict__ b0,
               const float* __restrict__ w1, const float* __restrict__ nb,
               float* __restrict__ x0out, float* __restrict__ x1out,
               int dopred, const float* __restrict__ predW, const float* __restrict__ predB,
               float* __restrict__ atomOut) {
  __shared__ float Q[C * SL];
  __shared__ float xs[C];
  const int a = blockIdx.x;
  const int tid = threadIdx.x;
  const int nslots = (kind == 0) ? 4 : 11;
  for (int i = tid; i < C * nslots; i += 128) {
    int ch = i / nslots, sl = i % nslots;
    float s = 0.f;
    for (int sp = 0; sp < nsplit; ++sp)
      s += P[(((long)sp * N_ATOMS + a) * C + ch) * SL + sl];
    Q[ch * SL + sl] = s;
  }
  __syncthreads();
  int g = tid;
  float y0, yv0 = 0.f, yv1 = 0.f, yv2 = 0.f;
  if (kind == 0) {
    y0 = b0[g];
    for (int f = 0; f < C; ++f) {
      y0 += w0[g * C + f] * Q[f * SL + 0];
      float w = w1[g * C + f];
      yv0 += w * Q[f * SL + 1]; yv1 += w * Q[f * SL + 2]; yv2 += w * Q[f * SL + 3];
    }
  } else {
    y0 = b0[g];
    for (int f = 0; f < C; ++f) {
      y0 += w0[g * 2 * C + f] * Q[f * SL + 0] + w0[g * 2 * C + C + f] * Q[f * SL + 1];
      float wa = w1[g * 3 * C + f], wb = w1[g * 3 * C + C + f], wc = w1[g * 3 * C + 2 * C + f];
      yv0 += wa * Q[f * SL + 2] + wb * Q[f * SL + 5] + wc * Q[f * SL + 8];
      yv1 += wa * Q[f * SL + 3] + wb * Q[f * SL + 6] + wc * Q[f * SL + 9];
      yv2 += wa * Q[f * SL + 4] + wb * Q[f * SL + 7] + wc * Q[f * SL + 10];
    }
  }
  float x0v = y0;
  if (kind != 2) {
    x0v = elu1(y0);
    float n = sqrtf(fmaxf(yv0 * yv0 + yv1 * yv1 + yv2 * yv2, 1e-12f));
    float gs = elu1(nb[g] + n) / n;
    yv0 *= gs; yv1 *= gs; yv2 *= gs;
  }
  x0out[a * C + g] = x0v;
  x1out[(a * C + g) * 3 + 0] = yv0;
  x1out[(a * C + g) * 3 + 1] = yv1;
  x1out[(a * C + g) * 3 + 2] = yv2;
  if (dopred) {
    xs[g] = x0v;
    __syncthreads();
    if (tid < A_TYPES) {
      float s = predB[tid];
      for (int gg = 0; gg < C; ++gg) s += xs[gg] * predW[tid * C + gg];
      atomOut[a * A_TYPES + tid] = s;
    }
  }
}

extern "C" void kernel_launch(void* const* d_in, const int* in_sizes, int n_in,
                              void* d_out, int out_size, void* d_ws, size_t ws_size,
                              hipStream_t stream) {
  (void)in_sizes; (void)n_in; (void)out_size;
  const float* r      = (const float*)d_in[0];
  const float* oneh   = (const float*)d_in[1];
  const float* embedW = (const float*)d_in[2];
  const float* embedB = (const float*)d_in[3];
  const float* Rw1    = (const float*)d_in[4];
  const float* Rb1    = (const float*)d_in[5];
  const float* Rw2    = (const float*)d_in[6];
  const float* Rb2    = (const float*)d_in[7];
  const float* si0w0  = (const float*)d_in[8];
  const float* si0b0  = (const float*)d_in[9];
  const float* si0w1  = (const float*)d_in[10];
  const float* si1w0  = (const float*)d_in[11];
  const float* si1b0  = (const float*)d_in[12];
  const float* si1w1  = (const float*)d_in[13];
  const float* si2w0  = (const float*)d_in[14];
  const float* si2b0  = (const float*)d_in[15];
  const float* si2w1  = (const float*)d_in[16];
  const float* nlb0   = (const float*)d_in[17];
  const float* nlb1   = (const float*)d_in[18];
  const float* predW  = (const float*)d_in[19];
  const float* predB  = (const float*)d_in[20];

  float* out = (float*)d_out;
  float* out_x0 = out;
  float* out_x1 = out + (size_t)N_ATOMS * C;
  float* out_at = out + (size_t)N_ATOMS * C * 4;

  float* ws = (float*)d_ws;
  size_t off = 0;
  float* e   = ws + off; off += (size_t)N_ATOMS * C;
  float* x0A = ws + off; off += (size_t)N_ATOMS * C;
  float* x1A = ws + off; off += (size_t)N_ATOMS * C * 3;
  float* x0B = ws + off; off += (size_t)N_ATOMS * C;
  float* x1B = ws + off; off += (size_t)N_ATOMS * C * 3;
  unsigned short* W2T = (unsigned short*)(ws + off); off += (size_t)12 * C * C / 2;
  float* P = ws + off;

  size_t wsFloats = ws_size / sizeof(float);
  size_t perSplit = (size_t)N_ATOMS * C * SL;
  int bsplit = 1;
  if (wsFloats > off + perSplit) {
    size_t avail = (wsFloats - off) / perSplit;
    bsplit = (int)(avail > 8 ? 8 : avail);
    if (bsplit < 1) bsplit = 1;
  }

  embed_kernel<<<(N_ATOMS * C + 255) / 256, 256, 0, stream>>>(oneh, embedW, embedB, e);
  w2t_kernel<<<(12 * C * C + 255) / 256, 256, 0, stream>>>(Rw2, W2T);

  dim3 pgrid(N_ATOMS / 16, bsplit);
  // layer 0 (filters 0-1, input = embedding)
  pair_kernel<0><<<pgrid, 256, 0, stream>>>(r, e, nullptr, Rw1, Rb1, W2T, Rb2, P, 0);
  si_kernel<<<N_ATOMS, 128, 0, stream>>>(P, bsplit, 0, si0w0, si0b0, si0w1, nlb0,
                                         x0A, x1A, 0, nullptr, nullptr, nullptr);
  // layer 1 (filters 2-6)
  pair_kernel<1><<<pgrid, 256, 0, stream>>>(r, x0A, x1A, Rw1, Rb1, W2T, Rb2, P, 2);
  si_kernel<<<N_ATOMS, 128, 0, stream>>>(P, bsplit, 1, si1w0, si1b0, si1w1, nlb1,
                                         x0B, x1B, 0, nullptr, nullptr, nullptr);
  // layer 2 (filters 7-11, linear) + prediction head
  pair_kernel<1><<<pgrid, 256, 0, stream>>>(r, x0B, x1B, Rw1, Rb1, W2T, Rb2, P, 7);
  si_kernel<<<N_ATOMS, 128, 0, stream>>>(P, bsplit, 2, si2w0, si2b0, si2w1, nullptr,
                                         out_x0, out_x1, 1, predW, predB, out_at);
}
